// DoseEncoder_5179730559269
// MI455X (gfx1250) — compile-verified, matrix operand order checked by reading, not run on hardware
//
#include <hip/hip_runtime.h>
#include <math.h>

// ---------------------------------------------------------------------------
// DoseEncoder on MI455X (gfx1250, wave32, WMMA f16 16x16x32).
//
// Transposed formulation per layer:  h_out^T[j,n] = sum_k W[k,j] * h_in^T[k,n]
//   A (16x32 f16)  = (W^T)/2 j-tile  (loop-invariant, in registers)
//   B (32x16 f16)  = activations^T   (K=hidden along VGPR/half, N=dose lane)
//   C/D (16x16 f32)= bias/2 / h^T/2  (N along lanes, M along VGPRs)
// The x0.5 of the tanh-form SiLU (silu(x) = h + h*tanh(h), h = x/2) is folded
// into the weight/bias fragments of layers 1-3, so each WMMA produces h
// directly and SiLU is just v_tanh_f32 + v_pk_fma_f32.
// D->B between layers is a per-lane f32->f16 pack (no shuffles, no LDS).
// ---------------------------------------------------------------------------

typedef __attribute__((ext_vector_type(16))) _Float16 v16h;
typedef __attribute__((ext_vector_type(8)))  float    v8f;

#define T_N   32768
#define D_N   256
#define HID   32
#define WPB   8            // waves per block
#define BLOCK (WPB * 32)

// K index held in {VGPR v, lane-half} for 16-bit A/B fragments (ISA 7.12.2):
// v0..3: K = 2v,2v+1 (lanes 0-15) / 2v+8,2v+9 (lanes 16-31); v4..7: +16.
__device__ __forceinline__ int kpat(int v, int half) {
    return 2 * (v & 3) + (half ? 8 : 0) + ((v & 4) ? 16 : 0);
}

// SiLU given h = x/2 (pre-scaled via weight folding):
// silu(x) = h + h*tanh(h)  ->  v_tanh_f32 + v_fma (packs to v_pk_fma_f32).
__device__ __forceinline__ float silu_h(float h) {
#if __has_builtin(__builtin_amdgcn_tanhf)
    return __builtin_fmaf(h, __builtin_amdgcn_tanhf(h), h);
#elif __has_builtin(__builtin_amdgcn_tanh_f32)
    return __builtin_fmaf(h, __builtin_amdgcn_tanh_f32(h), h);
#else
    float x = h + h;
    return x * __builtin_amdgcn_rcpf(1.0f + __expf(-x));
#endif
}

// Load scale*(W^T) j-tile as an A fragment: A[m][k] = scale*W[k][j0+m],
// zero-padded for k >= FI.
template <int FI>
__device__ __forceinline__ v16h load_A(const float* __restrict__ W,
                                       int j0, int m, int half, float scale) {
    v16h a;
#pragma unroll
    for (int v = 0; v < 8; ++v) {
        int k0 = kpat(v, half);
        a[2 * v]     = (k0     < FI) ? (_Float16)(scale * W[(k0)     * HID + j0 + m])
                                     : (_Float16)0.0f;
        a[2 * v + 1] = (k0 + 1 < FI) ? (_Float16)(scale * W[(k0 + 1) * HID + j0 + m])
                                     : (_Float16)0.0f;
    }
    return a;
}

// scale*bias as C fragment: C[r] = scale*b[j0 + r + 8*half].
__device__ __forceinline__ v8f load_C(const float* __restrict__ b, int j0, int half,
                                      float scale) {
    v8f c;
#pragma unroll
    for (int r = 0; r < 8; ++r) c[r] = scale * b[j0 + r + 8 * half];
    return c;
}

// SiLU + repack two f32 D fragments (already h-scaled; j=0..15, j=16..31)
// into next layer's f16 B fragment. Per-lane only.
__device__ __forceinline__ v16h pack_B(v8f lo, v8f hi) {
    v16h b;
#pragma unroll
    for (int e = 0; e < 8; ++e) {
        b[e]     = (_Float16)silu_h(lo[e]);
        b[e + 8] = (_Float16)silu_h(hi[e]);
    }
    return b;
}

__global__ void __launch_bounds__(BLOCK)
dose_encoder_wmma(const float* __restrict__ t_abs,
                  const float* __restrict__ dose_t,
                  const float* __restrict__ amts,
                  const float* __restrict__ ss,
                  const float* __restrict__ ii,
                  const float* __restrict__ span_p,
                  const float* __restrict__ log_sigma_p,
                  const float* __restrict__ W1, const float* __restrict__ b1,
                  const float* __restrict__ W2, const float* __restrict__ b2,
                  const float* __restrict__ W3, const float* __restrict__ b3,
                  const float* __restrict__ W4, const float* __restrict__ b4,
                  float* __restrict__ out) {
    __shared__ float s_dt[D_N], s_lamt[D_N], s_ssn[D_N], s_iin[D_N];

    const int   tid          = threadIdx.x;
    const float span         = span_p[0];
    const float inv_span     = 1.0f / span;
    const float inv_span_eps = 1.0f / (span + 1e-6f);
    const float inv_sigma    = __expf(-log_sigma_p[0]);   // 1/sigma

    // Stage per-dose metadata once per block (BLOCK == D_N).
    s_dt[tid]   = dose_t[tid];
    s_lamt[tid] = log1pf(amts[tid]);
    s_ssn[tid]  = ss[tid] * inv_span_eps;
    s_iin[tid]  = ii[tid] * inv_span_eps;
    __syncthreads();

    const int wave = tid >> 5;
    const int lane = tid & 31;
    const int m    = lane & 15;   // N (dose within chunk) / M (hidden-out row)
    const int half = lane >> 4;

    const int   trow = blockIdx.x * WPB + wave;   // one wave per observation t
    const float t    = t_abs[trow];

    // Loop-invariant weight A fragments + bias C fragments (registers).
    // Layers 1-3 are pre-scaled by 0.5 so WMMA emits h = x/2 for tanh-SiLU.
    const float HS = 0.5f;
    const v16h A1l = load_A<5>(W1, 0, m, half, HS),   A1h = load_A<5>(W1, 16, m, half, HS);
    const v16h A2l = load_A<HID>(W2, 0, m, half, HS), A2h = load_A<HID>(W2, 16, m, half, HS);
    const v16h A3l = load_A<HID>(W3, 0, m, half, HS), A3h = load_A<HID>(W3, 16, m, half, HS);
    const v8f  C1l = load_C(b1, 0, half, HS), C1h = load_C(b1, 16, half, HS);
    const v8f  C2l = load_C(b2, 0, half, HS), C2h = load_C(b2, 16, half, HS);
    const v8f  C3l = load_C(b3, 0, half, HS), C3h = load_C(b3, 16, half, HS);
    float w4a[8], w4b[8];
#pragma unroll
    for (int r = 0; r < 8; ++r) {
        w4a[r] = W4[r + 8 * half];
        w4b[r] = W4[16 + r + 8 * half];
    }
    const float bias4 = b4[0];

    // Time since last dose (masked min over all doses, wave reduction).
    float mn = 1e30f;
#pragma unroll
    for (int i = 0; i < D_N / 32; ++i) {
        float dti = (t - s_dt[lane + 32 * i]) * inv_span;
        mn = fminf(mn, dti >= 0.0f ? dti : 1e30f);
    }
#pragma unroll
    for (int off = 16; off > 0; off >>= 1) mn = fminf(mn, __shfl_xor(mn, off, 32));
    const float dt_last = mn;

    float acc = 0.0f;

    // Unroll x2: two independent WMMA/transcendental chains per iteration.
#pragma unroll 2
    for (int d0 = 0; d0 < D_N; d0 += 16) {
        const int   d   = d0 + m;
        const float dtv = (t - s_dt[d]) * inv_span;

        // Feature B fragment: lanes 0-15 carry K=0..4, everything else zero.
        v16h fb;
#pragma unroll
        for (int e = 0; e < 16; ++e) fb[e] = (_Float16)0.0f;
        fb[0] = half ? (_Float16)0.0f : (_Float16)dtv;
        fb[1] = half ? (_Float16)0.0f : (_Float16)s_lamt[d];
        fb[2] = half ? (_Float16)0.0f : (_Float16)dt_last;
        fb[3] = half ? (_Float16)0.0f : (_Float16)s_ssn[d];
        fb[4] = half ? (_Float16)0.0f : (_Float16)s_iin[d];

        // Layer 1 (5->32), bias via C; D = h1 = x1/2.
        v8f d1l = __builtin_amdgcn_wmma_f32_16x16x32_f16(false, A1l, false, fb,
                                                         (short)0, C1l, false, false);
        v8f d1h = __builtin_amdgcn_wmma_f32_16x16x32_f16(false, A1h, false, fb,
                                                         (short)0, C1h, false, false);
        v16h h1 = pack_B(d1l, d1h);

        // Layer 2 (32->32).
        v8f d2l = __builtin_amdgcn_wmma_f32_16x16x32_f16(false, A2l, false, h1,
                                                         (short)0, C2l, false, false);
        v8f d2h = __builtin_amdgcn_wmma_f32_16x16x32_f16(false, A2h, false, h1,
                                                         (short)0, C2h, false, false);
        v16h h2 = pack_B(d2l, d2h);

        // Layer 3 (32->32).
        v8f d3l = __builtin_amdgcn_wmma_f32_16x16x32_f16(false, A3l, false, h2,
                                                         (short)0, C3l, false, false);
        v8f d3h = __builtin_amdgcn_wmma_f32_16x16x32_f16(false, A3h, false, h2,
                                                         (short)0, C3h, false, false);

        // Layer 4 (32->1): per-lane dot over this lane's j subset, then add
        // the complementary half-lane's partial.
        float part = 0.0f;
#pragma unroll
        for (int r = 0; r < 8; ++r) {
            part += silu_h(d3l[r]) * w4a[r];
            part += silu_h(d3h[r]) * w4b[r];
        }
        part += __shfl_xor(part, 16, 32);
        const float score = part + bias4;

        // Masked Gaussian weight on normalized dt.
        const float x   = dtv * inv_sigma;
        const float wgt = (dtv >= 0.0f) ? __expf(-0.5f * x * x) : 0.0f;
        acc += (half == 0) ? score * wgt : 0.0f;   // avoid double count
    }

    // Sum the 16 per-dose-lane partials -> out[t].
#pragma unroll
    for (int off = 16; off > 0; off >>= 1) acc += __shfl_xor(acc, off, 32);
    if (lane == 0) out[trow] = acc;
}

extern "C" void kernel_launch(void* const* d_in, const int* in_sizes, int n_in,
                              void* d_out, int out_size, void* d_ws, size_t ws_size,
                              hipStream_t stream) {
    (void)in_sizes; (void)n_in; (void)out_size; (void)d_ws; (void)ws_size;
    const float* t_abs  = (const float*)d_in[0];
    const float* dose_t = (const float*)d_in[1];
    const float* amts   = (const float*)d_in[2];
    const float* ss     = (const float*)d_in[3];
    const float* ii     = (const float*)d_in[4];
    const float* span_p = (const float*)d_in[5];
    const float* lsig   = (const float*)d_in[6];
    const float* W1 = (const float*)d_in[7];  const float* b1 = (const float*)d_in[8];
    const float* W2 = (const float*)d_in[9];  const float* b2 = (const float*)d_in[10];
    const float* W3 = (const float*)d_in[11]; const float* b3 = (const float*)d_in[12];
    const float* W4 = (const float*)d_in[13]; const float* b4 = (const float*)d_in[14];
    float* out = (float*)d_out;

    dim3 grid(T_N / WPB);   // one wave32 per observation time
    dim3 block(BLOCK);
    dose_encoder_wmma<<<grid, block, 0, stream>>>(
        t_abs, dose_t, amts, ss, ii, span_p, lsig,
        W1, b1, W2, b2, W3, b3, W4, b4, out);
}
